// GRUWorldModel_36172214567609
// MI455X (gfx1250) — compile-verified, matrix operand order checked by reading, not run on hardware
//
#include <hip/hip_runtime.h>
#include <hip/hip_bf16.h>

// ---------------------------------------------------------------------------
// GRU world model for MI455X (gfx1250): persistent wave32 WMMA kernel, v2.
//   - bf16 A/B, f32 accumulate via v_wmma_f32_16x16x32_bf16
//   - weight fragments staged global -> LDS (double-buffered, 8x traffic cut:
//     all 8 waves of a WG share the same B fragments)
//   - #pragma unroll 1 on K-chunk and time loops: keeps the steady-state body
//     small so nothing spills to scratch (round-1 pathology)
//   - one persistent kernel, software grid barrier between recurrence phases
// ---------------------------------------------------------------------------

typedef __bf16 bf16_t;
typedef __attribute__((ext_vector_type(16))) __bf16 v16bf;
typedef __attribute__((ext_vector_type(8)))  __bf16 v8bf;
typedef __attribute__((ext_vector_type(8)))  float  v8f;

#define BATCH 128
#define HID   1024
#define LAT   512
#define TIN   64
#define TOUT  64
#define NWG   32     // persistent workgroups; each owns 32 hidden columns
#define TNC   32     // hidden columns per WG
#define NTHR  256    // 8 wave32; wave w owns batch rows [16w,16w+16)
#define KCH   4      // K-steps (of 32) staged per LDS chunk
#define RMAX  6      // max fragment runs (3 gates x 2 n-tiles)
#define STAGE_ELEMS (RMAX * KCH * 512)   // bf16 elems per LDS buffer (24KB)

// ---------------- fragment helpers -----------------------------------------

// A fragment (16x32 bf16): lane l -> row m0+(l&15),
// K chunks [kb..kb+7], [kb+16..kb+23], kb = k0 + 8*(l>=16)
__device__ __forceinline__ v16bf ld_a(const bf16_t* __restrict__ X, int ld,
                                      int m0, int k0, int lane) {
  const bf16_t* p = X + (size_t)(m0 + (lane & 15)) * ld + (k0 + ((lane >> 4) << 3));
  v8bf lo = *(const v8bf*)(p);
  v8bf hi = *(const v8bf*)(p + 16);
  return __builtin_shufflevector(lo, hi, 0,1,2,3,4,5,6,7,8,9,10,11,12,13,14,15);
}

// B fragment from a packed 512-elem tile (works for global or LDS pointers):
// half0 at +lane*8, half1 at +256+lane*8 -> two b128 per lane, bank-conflict-free.
__device__ __forceinline__ v16bf ld_frag(const bf16_t* __restrict__ T, int frag,
                                         int lane) {
  const bf16_t* p = T + (size_t)frag * 512 + lane * 8;
  v8bf lo = *(const v8bf*)(p);
  v8bf hi = *(const v8bf*)(p + 256);
  return __builtin_shufflevector(lo, hi, 0,1,2,3,4,5,6,7,8,9,10,11,12,13,14,15);
}

__device__ __forceinline__ v8f wmma_bf(v16bf a, v16bf b, v8f c) {
  return __builtin_amdgcn_wmma_f32_16x16x32_bf16(false, a, false, b, (short)0, c,
                                                 false, false);
}

__device__ __forceinline__ float sigm(float x) { return 1.f / (1.f + __expf(-x)); }

// ---------------- grid-wide barrier ----------------------------------------

__device__ __forceinline__ void grid_sync(int* bar, int& target) {
  __syncthreads();
  target += NWG;
  if (threadIdx.x == 0) {
    __threadfence();
    __hip_atomic_fetch_add(bar, 1, __ATOMIC_ACQ_REL, __HIP_MEMORY_SCOPE_AGENT);
    while (__hip_atomic_load(bar, __ATOMIC_ACQUIRE, __HIP_MEMORY_SCOPE_AGENT) < target)
      __builtin_amdgcn_s_sleep(2);
  }
  __syncthreads();
}

// ---------------- LDS staging of weight fragments --------------------------
// Copy R runs x KCH tiles (512 bf16 each) into LDS. KCH*64 == NTHR, so each
// thread moves exactly one 16B chunk per run (global_load_b128 + ds_store_b128).

template <int R>
__device__ __forceinline__ void stage(const bf16_t* __restrict__ Wp,
                                      const int* tbase, int kt0,
                                      bf16_t* __restrict__ dst, int tid) {
#pragma unroll
  for (int r = 0; r < R; ++r) {
    const bf16_t* src = Wp + (size_t)(tbase[r] + kt0) * 512;
    bf16_t* d = dst + r * (KCH * 512);
    for (int c = tid; c < KCH * 64; c += NTHR)
      *(v8bf*)(d + c * 8) = *(const v8bf*)(src + c * 8);
  }
}

// ---------------- staged multi-run GEMM ------------------------------------
// acc[r] += A(m0-tile) x B(run r), K-loop double-buffered through LDS.

template <int R>
__device__ __forceinline__ void gemm_staged(
    const bf16_t* __restrict__ X, int ldx, int K,
    const bf16_t* __restrict__ Wp, const int* tbase,
    int m0, int lane, bf16_t* __restrict__ lds, v8f* acc) {
  const int kts = K >> 5;
  const int tid = threadIdx.x;
  int par = 0;
  stage<R>(Wp, tbase, 0, lds, tid);
  __syncthreads();
#pragma unroll 1
  for (int kc = 0; kc < kts; kc += KCH) {
    if (kc + KCH < kts)
      stage<R>(Wp, tbase, kc + KCH, lds + (par ^ 1) * STAGE_ELEMS, tid);
    const bf16_t* buf = lds + par * STAGE_ELEMS;
#pragma unroll
    for (int q = 0; q < KCH; ++q) {
      v16bf a = ld_a(X, ldx, m0, (kc + q) << 5, lane);
#pragma unroll
      for (int r = 0; r < R; ++r) {
        v16bf b = ld_frag(buf, r * KCH + q, lane);
        acc[r] = wmma_bf(a, b, acc[r]);
      }
    }
    __syncthreads();
    par ^= 1;
  }
}

// ---------------- fused GRU cell phase -------------------------------------
// Six gate tiles (r,z,n x 2 n-tiles) for batch rows [m0,m0+16),
// hidden cols [n0,n0+32); then the elementwise GRU update in registers.

__device__ __forceinline__ void gru_layer(
    const bf16_t* __restrict__ X, int ldx, int Kx,
    const bf16_t* __restrict__ Hpb, const float* __restrict__ Hpf,
    const bf16_t* __restrict__ Wih, const bf16_t* __restrict__ Whh,
    const float* __restrict__ bih, const float* __restrict__ bhh,
    float* __restrict__ Hnf, bf16_t* __restrict__ Hnb,
    int n0, int m0, int lane, bf16_t* __restrict__ lds) {
  v8f aI[6] = {};
  v8f aH[6] = {};
  int tb[6];
  const int ktI = Kx >> 5;
#pragma unroll
  for (int g = 0; g < 3; ++g)
#pragma unroll
    for (int j = 0; j < 2; ++j)
      tb[g * 2 + j] = (((g * HID + n0) >> 4) + j) * ktI;
  gemm_staged<6>(X, ldx, Kx, Wih, tb, m0, lane, lds, aI);

  const int ktH = HID >> 5;
#pragma unroll
  for (int g = 0; g < 3; ++g)
#pragma unroll
    for (int j = 0; j < 2; ++j)
      tb[g * 2 + j] = (((g * HID + n0) >> 4) + j) * ktH;
  gemm_staged<6>(Hpb, HID, HID, Whh, tb, m0, lane, lds, aH);

  // D layout: VGPR r -> M = m0 + r + 8*(lane>=16), N = lane&15
  const int mb = m0 + ((lane >> 4) << 3);
  const int nl = lane & 15;
#pragma unroll
  for (int j = 0; j < 2; ++j) {
    const int n = n0 + j * 16 + nl;
    float br[3], bh[3];
#pragma unroll
    for (int g = 0; g < 3; ++g) {
      br[g] = bih[g * HID + n];
      bh[g] = bhh[g * HID + n];
    }
#pragma unroll
    for (int r8 = 0; r8 < 8; ++r8) {
      const int m = mb + r8;
      const float rg = sigm((aI[0 + j][r8] + br[0]) + (aH[0 + j][r8] + bh[0]));
      const float zg = sigm((aI[2 + j][r8] + br[1]) + (aH[2 + j][r8] + bh[1]));
      const float ng = tanhf((aI[4 + j][r8] + br[2]) + rg * (aH[4 + j][r8] + bh[2]));
      const float hp = Hpf[(size_t)m * HID + n];
      const float hn = (1.f - zg) * ng + zg * hp;
      Hnf[(size_t)m * HID + n] = hn;
      Hnb[(size_t)m * HID + n] = (bf16_t)hn;
    }
  }
}

// ---------------- persistent world-model kernel ----------------------------

__global__ __launch_bounds__(NTHR, 1) void gru_worldmodel_persistent(
    const bf16_t* __restrict__ xseq, bf16_t* __restrict__ xcur,
    const bf16_t* __restrict__ Wih0, const bf16_t* __restrict__ Whh0,
    const bf16_t* __restrict__ Wih1, const bf16_t* __restrict__ Whh1,
    const bf16_t* __restrict__ W1p, const bf16_t* __restrict__ W2p,
    const float* __restrict__ bih0, const float* __restrict__ bhh0,
    const float* __restrict__ bih1, const float* __restrict__ bhh1,
    const float* __restrict__ bp1, const float* __restrict__ bp2,
    float* __restrict__ h0f, bf16_t* __restrict__ h0b,
    float* __restrict__ h1f, bf16_t* __restrict__ h1b,
    bf16_t* __restrict__ pbuf, float* __restrict__ out, int* bar) {
  __shared__ bf16_t lds_stage[2 * STAGE_ELEMS];   // 48 KB double buffer

  const int ng = blockIdx.x;
  const int lane = threadIdx.x & 31;
  const int m0 = (threadIdx.x >> 5) << 4;
  const int n0 = ng * TNC;
  int bt = 0;

  const size_t HSZ = (size_t)BATCH * HID;

  // ---- encoder: 64 steps x (L0, L1) ----
#pragma unroll 1
  for (int t = 0; t < TIN; ++t) {
    const int pr = t & 1, pw = pr ^ 1;
    gru_layer(xseq + (size_t)t * LAT, TIN * LAT, LAT,
              h0b + pr * HSZ, h0f + pr * HSZ, Wih0, Whh0, bih0, bhh0,
              h0f + pw * HSZ, h0b + pw * HSZ, n0, m0, lane, lds_stage);
    grid_sync(bar, bt);
    gru_layer(h0b + pw * HSZ, HID, HID,
              h1b + pr * HSZ, h1f + pr * HSZ, Wih1, Whh1, bih1, bhh1,
              h1f + pw * HSZ, h1b + pw * HSZ, n0, m0, lane, lds_stage);
    grid_sync(bar, bt);
  }

  // ---- decoder: 64 steps x (L0, L1, proj1, proj2) ----
#pragma unroll 1
  for (int t = 0; t < TOUT; ++t) {
    const int s = TIN + t;
    const int pr = s & 1, pw = pr ^ 1;
    gru_layer(xcur, LAT, LAT,
              h0b + pr * HSZ, h0f + pr * HSZ, Wih0, Whh0, bih0, bhh0,
              h0f + pw * HSZ, h0b + pw * HSZ, n0, m0, lane, lds_stage);
    grid_sync(bar, bt);
    gru_layer(h0b + pw * HSZ, HID, HID,
              h1b + pr * HSZ, h1f + pr * HSZ, Wih1, Whh1, bih1, bhh1,
              h1f + pw * HSZ, h1b + pw * HSZ, n0, m0, lane, lds_stage);
    grid_sync(bar, bt);
    // p = relu(h1 @ w1^T + bp1)
    {
      v8f acc[2] = {};
      int tb[2];
      const int kts = HID >> 5;
      tb[0] = (n0 >> 4) * kts;
      tb[1] = ((n0 >> 4) + 1) * kts;
      gemm_staged<2>(h1b + pw * HSZ, HID, HID, W1p, tb, m0, lane, lds_stage, acc);
      const int mb = m0 + ((lane >> 4) << 3);
#pragma unroll
      for (int j = 0; j < 2; ++j) {
        const int n = n0 + j * 16 + (lane & 15);
        const float bb = bp1[n];
#pragma unroll
        for (int r8 = 0; r8 < 8; ++r8) {
          float v = acc[j][r8] + bb;
          v = v > 0.f ? v : 0.f;
          pbuf[(size_t)(mb + r8) * HID + n] = (bf16_t)v;
        }
      }
    }
    grid_sync(bar, bt);
    // y = p @ w2^T + bp2  (N = 512 -> first 16 WGs compute)
    if (ng < 16) {
      v8f acc[2] = {};
      int tb[2];
      const int kts = HID >> 5;
      tb[0] = (n0 >> 4) * kts;
      tb[1] = ((n0 >> 4) + 1) * kts;
      gemm_staged<2>(pbuf, HID, HID, W2p, tb, m0, lane, lds_stage, acc);
      const int mb = m0 + ((lane >> 4) << 3);
#pragma unroll
      for (int j = 0; j < 2; ++j) {
        const int n = n0 + j * 16 + (lane & 15);
        const float bb = bp2[n];
#pragma unroll
        for (int r8 = 0; r8 < 8; ++r8) {
          const float y = acc[j][r8] + bb;
          const int m = mb + r8;
          out[((size_t)m * TOUT + t) * LAT + n] = y;   // (B, T, LAT)
          xcur[(size_t)m * LAT + n] = (bf16_t)y;       // next-step input
        }
      }
    }
    grid_sync(bar, bt);
  }
}

// ---------------- prep kernels ---------------------------------------------

__global__ void k_init_state(float* h0f, float* h1f, bf16_t* h0b, bf16_t* h1b,
                             int* bar) {
  const size_t i = (size_t)blockIdx.x * blockDim.x + threadIdx.x;
  const size_t n = 2 * (size_t)BATCH * HID;
  if (i < n) {
    h0f[i] = 0.f; h1f[i] = 0.f;
    h0b[i] = (bf16_t)0.f; h1b[i] = (bf16_t)0.f;
  }
  if (i == 0) *bar = 0;
}

__global__ void k_cvt_x(const float* __restrict__ ctx, bf16_t* __restrict__ xseq,
                        bf16_t* __restrict__ xcur) {
  const size_t i = (size_t)blockIdx.x * blockDim.x + threadIdx.x;
  const size_t n = (size_t)BATCH * TIN * LAT;
  if (i < n) xseq[i] = (bf16_t)ctx[i];
  if (i < (size_t)BATCH * LAT) {
    const size_t b = i / LAT, k = i % LAT;
    xcur[i] = (bf16_t)ctx[(b * TIN + (TIN - 1)) * LAT + k];  // x0 = last context
  }
}

// Pack row-major f32 weight (Nr x K) into bf16 WMMA B-fragment tiles:
// tile (nt,kt) -> 512 bf16; lane l holds col n = nt*16+(l&15),
// K chunks kb..kb+7 (half0) and kb+16..kb+23 (half1), kb = kt*32 + 8*(l>=16).
__global__ void k_pack_w(const float* __restrict__ W, bf16_t* __restrict__ dst,
                         int Nr, int K) {
  const int idx = blockIdx.x * blockDim.x + threadIdx.x;
  const int ktiles = K >> 5;
  const int chunks = (Nr >> 4) * ktiles * 64;
  if (idx >= chunks) return;
  const int half = idx & 1;
  const int lane = (idx >> 1) & 31;
  const int tile = idx >> 6;
  const int nt = tile / ktiles, kt = tile % ktiles;
  const int n = nt * 16 + (lane & 15);
  const int k0 = kt * 32 + ((lane >> 4) << 3) + (half ? 16 : 0);
  const float* s = W + (size_t)n * K + k0;
  bf16_t* d = dst + (size_t)tile * 512 + half * 256 + lane * 8;
#pragma unroll
  for (int i = 0; i < 8; ++i) d[i] = (bf16_t)s[i];
}

// ---------------- launch ---------------------------------------------------

extern "C" void kernel_launch(void* const* d_in, const int* in_sizes, int n_in,
                              void* d_out, int out_size, void* d_ws, size_t ws_size,
                              hipStream_t stream) {
  const float* ctx  = (const float*)d_in[0];
  const float* wih0 = (const float*)d_in[1];
  const float* whh0 = (const float*)d_in[2];
  const float* bih0 = (const float*)d_in[3];
  const float* bhh0 = (const float*)d_in[4];
  const float* wih1 = (const float*)d_in[5];
  const float* whh1 = (const float*)d_in[6];
  const float* bih1 = (const float*)d_in[7];
  const float* bhh1 = (const float*)d_in[8];
  const float* w1   = (const float*)d_in[9];
  const float* bp1  = (const float*)d_in[10];
  const float* w2   = (const float*)d_in[11];
  const float* bp2  = (const float*)d_in[12];
  float* out = (float*)d_out;

  char* ws = (char*)d_ws;
  size_t off = 0;
  auto take = [&](size_t bytes) -> char* {
    char* p = ws + off;
    off = (off + bytes + 255) & ~(size_t)255;
    return p;
  };
  int*    bar   = (int*)take(256);
  bf16_t* Wih0p = (bf16_t*)take((size_t)3072 * 512 * 2);
  bf16_t* Whh0p = (bf16_t*)take((size_t)3072 * 1024 * 2);
  bf16_t* Wih1p = (bf16_t*)take((size_t)3072 * 1024 * 2);
  bf16_t* Whh1p = (bf16_t*)take((size_t)3072 * 1024 * 2);
  bf16_t* W1p   = (bf16_t*)take((size_t)1024 * 1024 * 2);
  bf16_t* W2p   = (bf16_t*)take((size_t)512 * 1024 * 2);
  bf16_t* xseq  = (bf16_t*)take((size_t)BATCH * TIN * LAT * 2);
  bf16_t* xcur  = (bf16_t*)take((size_t)BATCH * LAT * 2);
  float*  h0f   = (float*)take((size_t)2 * BATCH * HID * 4);
  float*  h1f   = (float*)take((size_t)2 * BATCH * HID * 4);
  bf16_t* h0b   = (bf16_t*)take((size_t)2 * BATCH * HID * 2);
  bf16_t* h1b   = (bf16_t*)take((size_t)2 * BATCH * HID * 2);
  bf16_t* pbuf  = (bf16_t*)take((size_t)BATCH * HID * 2);

  {
    const int n = 2 * BATCH * HID;
    k_init_state<<<(n + 255) / 256, 256, 0, stream>>>(h0f, h1f, h0b, h1b, bar);
  }
  {
    const size_t n = (size_t)BATCH * TIN * LAT;
    k_cvt_x<<<(unsigned)((n + 255) / 256), 256, 0, stream>>>(ctx, xseq, xcur);
  }
  auto pack = [&](const float* src, bf16_t* dst, int Nr, int K) {
    const int chunks = (Nr >> 4) * (K >> 5) * 64;
    k_pack_w<<<(chunks + 255) / 256, 256, 0, stream>>>(src, dst, Nr, K);
  };
  pack(wih0, Wih0p, 3072, 512);
  pack(whh0, Whh0p, 3072, 1024);
  pack(wih1, Wih1p, 3072, 1024);
  pack(whh1, Whh1p, 3072, 1024);
  pack(w1,   W1p,   1024, 1024);
  pack(w2,   W2p,    512, 1024);

  gru_worldmodel_persistent<<<NWG, NTHR, 0, stream>>>(
      xseq, xcur, Wih0p, Whh0p, Wih1p, Whh1p, W1p, W2p,
      bih0, bhh0, bih1, bhh1, bp1, bp2,
      h0f, h0b, h1f, h1b, pbuf, out, bar);
}
